// MultiHeadAttention_70781061038852
// MI455X (gfx1250) — matrix operand order repaired and checked by reading, not compiled, not run on hardware
//
#include <hip/hip_runtime.h>

#define DEV_INLINE __device__ __forceinline__

#ifndef __has_builtin
#define __has_builtin(x) 0
#endif

#if __has_builtin(__builtin_amdgcn_global_load_async_to_lds_b128)
#define HAVE_ASYNC_LDS 1
#else
#define HAVE_ASYNC_LDS 0
#endif

typedef __attribute__((ext_vector_type(16))) __bf16 v16bf;
typedef __attribute__((ext_vector_type(8)))  float  v8f;

typedef int v4i_vec __attribute__((vector_size(16)));
#if HAVE_ASYNC_LDS
// Builtin prototype (from hipcc diagnostic): arg0 = AS1 (global) int4*,
// arg1 = AS3 (LDS) int4*, arg2/arg3 = imm offset / cpol.
typedef __attribute__((address_space(1))) v4i_vec* gptr_b128;
typedef __attribute__((address_space(3))) v4i_vec* lptr_b128;
#endif

namespace {

constexpr int kB  = 2;
constexpr int kS  = 2048;
constexpr int kD  = 1024;
constexpr int kH  = 16;
constexpr int kDK = 64;          // kD / kH

constexpr int TM = 64;           // workgroup M tile
constexpr int TN = 64;           // workgroup N tile
constexpr int TK = 32;           // K step == WMMA bf16 K
constexpr int THREADS = 128;     // 4 waves (wave32)
constexpr int LDSPAD = 40;       // 32 + 8 ushort pad (80B rows, 16B aligned)

DEV_INLINE unsigned short f32_to_bf16(float f) {
  union { float f; unsigned u; } c; c.f = f;
  unsigned r = c.u + 0x7FFFu + ((c.u >> 16) & 1u);   // round-to-nearest-even
  return (unsigned short)(r >> 16);
}

DEV_INLINE unsigned pack_bf16x2(float a, float b) {
  return (unsigned)f32_to_bf16(a) | ((unsigned)f32_to_bf16(b) << 16);
}

// 16B global -> 16B LDS, async (ASYNCcnt) if the CDNA5 path is available.
DEV_INLINE void copy_b128_to_lds(const unsigned short* g, unsigned short* l) {
#if HAVE_ASYNC_LDS
  __builtin_amdgcn_global_load_async_to_lds_b128(
      (gptr_b128)g, (lptr_b128)l, /*offset=*/0, /*cpol=*/0);
#else
  *(uint4*)l = *(const uint4*)g;
#endif
}

DEV_INLINE void wait_async_lds() {
#if HAVE_ASYNC_LDS
#if __has_builtin(__builtin_amdgcn_s_wait_asynccnt)
  __builtin_amdgcn_s_wait_asynccnt(0);
#else
  asm volatile("s_wait_asynccnt 0x0" ::: "memory");
#endif
#endif
}

union Frag16 { v16bf v; unsigned short s[16]; };

// A-matrix 16x32 bf16 fragment (ISA 7.12.2): lane m = row; elements 0..7 hold
// K = 8*g + 0..7, elements 8..15 hold K = 16 + 8*g + 0..7  (g = lane>>4).
DEV_INLINE v16bf load_a_frag(const unsigned short* lds, int row, int lane) {
  Frag16 f;
  const int g = lane >> 4;
  const unsigned short* p = lds + row * LDSPAD;
#pragma unroll
  for (int i = 0; i < 8; ++i) {
    f.s[i]     = p[8 * g + i];
    f.s[i + 8] = p[16 + 8 * g + i];
  }
  return f.v;
}

// B-matrix 32x16 bf16 fragment: lane n = col (lane&15); elements 0..7 hold
// K = 8*g + 0..7, elements 8..15 hold K = 16 + 8*g + 0..7. LDS tile n-major [n][k].
DEV_INLINE v16bf load_b_frag(const unsigned short* lds, int col, int lane) {
  Frag16 f;
  const int g = lane >> 4;
  const unsigned short* p = lds + col * LDSPAD;
#pragma unroll
  for (int i = 0; i < 8; ++i) {
    f.s[i]     = p[8 * g + i];
    f.s[i + 8] = p[16 + 8 * g + i];
  }
  return f.v;
}

DEV_INLINE v8f wmma_bf16(v16bf a, v16bf b, v8f c) {
  return __builtin_amdgcn_wmma_f32_16x16x32_bf16(
      /*neg_a=*/false, a, /*neg_b=*/false, b,
      /*c_mod=*/(short)0, c, /*reuse_a=*/false, /*reuse_b=*/false);
}

} // namespace

// ---------------------------------------------------------------------------
// Streaming fp32 -> bf16 (float4 -> packed uint2).
// ---------------------------------------------------------------------------
__global__ __launch_bounds__(256)
void convert_bf16_kernel(const float4* __restrict__ in, uint2* __restrict__ out,
                         int n4) {
  const int i = blockIdx.x * blockDim.x + threadIdx.x;
  if (i < n4) {
    const float4 v = in[i];
    uint2 o;
    o.x = pack_bf16x2(v.x, v.y);
    o.y = pack_bf16x2(v.z, v.w);
    out[i] = o;
  }
}

// ---------------------------------------------------------------------------
// WT[n][k] = bf16(W[k][n])  for 1024x1024 weights. grid (32,32), block (32,8)
// ---------------------------------------------------------------------------
__global__ __launch_bounds__(256)
void transpose_convert_kernel(const float* __restrict__ W,
                              unsigned short* __restrict__ WT) {
  __shared__ float tile[32][33];
  const int c0 = blockIdx.x * 32, r0 = blockIdx.y * 32;
  const int tx = threadIdx.x;
#pragma unroll
  for (int i = threadIdx.y; i < 32; i += 8)
    tile[i][tx] = W[(size_t)(r0 + i) * kD + c0 + tx];
  __syncthreads();
#pragma unroll
  for (int i = threadIdx.y; i < 32; i += 8)
    WT[(size_t)(c0 + i) * kD + r0 + tx] = f32_to_bf16(tile[tx][i]);
}

// ---------------------------------------------------------------------------
// C = A[M,K] @ B^T  with A bf16 row-major (lda), B bf16 stored [N][K] (ldb),
// bias fp32. MODE 0: bf16 row-major out (ldc). MODE 1: bf16 V^T out [B][D][S]
// (packed b128 stores). MODE 2: fp32 row-major out. grid: (N/TN, M/TM).
// ---------------------------------------------------------------------------
template <int MODE>
__global__ __launch_bounds__(THREADS)
void gemm_bf16_kernel(const unsigned short* __restrict__ A,
                      const unsigned short* __restrict__ Bw,
                      const float* __restrict__ bias, void* __restrict__ Cout,
                      int K, int lda, int ldb, int ldc) {
  __shared__ __align__(16) unsigned short lA[TM * LDSPAD];
  __shared__ __align__(16) unsigned short lB[TN * LDSPAD];

  const int tid  = threadIdx.x;
  const int lane = tid & 31;
  const int wave = tid >> 5;
  const int wm = (wave >> 1) * 32;
  const int wn = (wave & 1) * 32;
  const int row0 = blockIdx.y * TM;
  const int col0 = blockIdx.x * TN;

  const v8f zero = {0.f, 0.f, 0.f, 0.f, 0.f, 0.f, 0.f, 0.f};
  v8f acc[2][2];
  acc[0][0] = zero; acc[0][1] = zero; acc[1][0] = zero; acc[1][1] = zero;

  for (int kt = 0; kt < K; kt += TK) {
    // Both tiles are 64 rows x 32 bf16 (64B rows): 4x b128 chunks per row,
    // 256 chunk-copies each -> 2 per thread, pure async traffic.
#pragma unroll
    for (int e = 0; e < 2; ++e) {
      const int o = tid + e * THREADS;              // 0..255
      const int r = o >> 2, ch = o & 3;
      copy_b128_to_lds(A  + (size_t)(row0 + r) * lda + kt + ch * 8,
                       &lA[r * LDSPAD + ch * 8]);
      copy_b128_to_lds(Bw + (size_t)(col0 + r) * ldb + kt + ch * 8,
                       &lB[r * LDSPAD + ch * 8]);
    }
    wait_async_lds();
    __syncthreads();

    v16bf af[2], bfr[2];
#pragma unroll
    for (int r = 0; r < 2; ++r)
      af[r] = load_a_frag(lA, wm + r * 16 + (lane & 15), lane);
#pragma unroll
    for (int c = 0; c < 2; ++c)
      bfr[c] = load_b_frag(lB, wn + c * 16 + (lane & 15), lane);
#pragma unroll
    for (int r = 0; r < 2; ++r)
#pragma unroll
      for (int c = 0; c < 2; ++c)
        acc[r][c] = wmma_bf16(af[r], bfr[c], acc[r][c]);
    __syncthreads();
  }

  const int g = lane >> 4, n15 = lane & 15;
#pragma unroll
  for (int r = 0; r < 2; ++r)
#pragma unroll
    for (int c = 0; c < 2; ++c) {
      const int col = col0 + wn + c * 16 + n15;
      const float bv = bias[col];
      if constexpr (MODE == 1) {
        // V^T bf16 [b][d][s]: this lane's 8 values are contiguous along s.
        const int rowBase = row0 + wm + r * 16;       // multiple of 16
        const int b = rowBase >> 11;
        const int s = (rowBase & (kS - 1)) + 8 * g;   // 8-aligned
        union { unsigned short s8[8]; uint4 v; } pk;
#pragma unroll
        for (int vr = 0; vr < 8; ++vr)
          pk.s8[vr] = f32_to_bf16(acc[r][c][vr] + bv);
        *(uint4*)&((unsigned short*)Cout)[((size_t)b * kD + col) * kS + s] =
            pk.v;
      } else {
#pragma unroll
        for (int vr = 0; vr < 8; ++vr) {
          const int row = row0 + wm + r * 16 + vr + 8 * g;
          const float val = acc[r][c][vr] + bv;
          if constexpr (MODE == 0)
            ((unsigned short*)Cout)[(size_t)row * ldc + col] = f32_to_bf16(val);
          else
            ((float*)Cout)[(size_t)row * ldc + col] = val;
        }
      }
    }
}

// ---------------------------------------------------------------------------
// scores[bh,q,n] = mask ? (Q_head[q,:]·K_head[n,:]) / 8 : -1e9
// Q,K are bf16 [B*S][D]; both operands k-contiguous -> async tiles.
// grid: (S/TN, S/TM, B*H)
// ---------------------------------------------------------------------------
__global__ __launch_bounds__(THREADS)
void attn_scores_kernel(const unsigned short* __restrict__ Qb,
                        const unsigned short* __restrict__ Kb,
                        const int* __restrict__ mask, float* __restrict__ attn) {
  __shared__ __align__(16) unsigned short lA[TM * LDSPAD];
  __shared__ __align__(16) unsigned short lB[TN * LDSPAD];

  const int tid  = threadIdx.x;
  const int lane = tid & 31;
  const int wave = tid >> 5;
  const int wm = (wave >> 1) * 32;
  const int wn = (wave & 1) * 32;
  const int qt = blockIdx.y * TM;
  const int nt = blockIdx.x * TN;
  const int bh = blockIdx.z;
  const int b = bh / kH, h = bh % kH;
  const size_t bRow = (size_t)b * kS;
  const int hOff = h * kDK;
  float* out = attn + (size_t)bh * kS * kS;

  const v8f zero = {0.f, 0.f, 0.f, 0.f, 0.f, 0.f, 0.f, 0.f};
  v8f acc[2][2];
  acc[0][0] = zero; acc[0][1] = zero; acc[1][0] = zero; acc[1][1] = zero;

  for (int kt = 0; kt < kDK; kt += TK) {               // 2 iterations
    // 64 rows x 16 bf16 (k = kt..kt+15) per tile -> 128 b128 chunks -> 1 per thread per tile
    const int r = tid >> 1, ch = tid & 1;
    copy_b128_to_lds(Qb + (bRow + qt + r) * kD + hOff + kt + ch * 8,
                     &lA[r * LDSPAD + ch * 8]);
    copy_b128_to_lds(Kb + (bRow + nt + r) * kD + hOff + kt + ch * 8,
                     &lB[r * LDSPAD + ch * 8]);
    wait_async_lds();
    __syncthreads();

    v16bf af[2], bfr[2];
#pragma unroll
    for (int r2 = 0; r2 < 2; ++r2)
      af[r2] = load_a_frag(lA, wm + r2 * 16 + (lane & 15), lane);
#pragma unroll
    for (int c = 0; c < 2; ++c)
      bfr[c] = load_b_frag(lB, wn + c * 16 + (lane & 15), lane);
#pragma unroll
    for (int r2 = 0; r2 < 2; ++r2)
#pragma unroll
      for (int c = 0; c < 2; ++c)
        acc[r2][c] = wmma_bf16(af[r2], bfr[c], acc[r2][c]);
    __syncthreads();
  }

  const int g = lane >> 4, n15 = lane & 15;
  const float scale = 0.125f;                           // 1/sqrt(64)
#pragma unroll
  for (int r = 0; r < 2; ++r)
#pragma unroll
    for (int c = 0; c < 2; ++c) {
      const int col = nt + wn + c * 16 + n15;
#pragma unroll
      for (int vr = 0; vr < 8; ++vr) {
        const int row = qt + wm + r * 16 + vr + 8 * g;
        const float s = acc[r][c][vr] * scale;
        const int mv = mask[(size_t)row * kS + col];    // 16MB (S*S int32), L2-resident
        out[(size_t)row * kS + col] = mv ? s : -1e9f;
      }
    }
}

// ---------------------------------------------------------------------------
// Row softmax in place, one wave32 per row of 2048. grid: B*H*S/4, block 128
// ---------------------------------------------------------------------------
__global__ __launch_bounds__(THREADS)
void softmax_kernel(float* __restrict__ attn) {
  const int lane = threadIdx.x & 31;
  const int wave = threadIdx.x >> 5;
  const size_t row = (size_t)blockIdx.x * 4 + wave;
  float* p = attn + row * kS;

  float m = -3.0e38f;
  for (int c = lane; c < kS; c += 32) m = fmaxf(m, p[c]);
#pragma unroll
  for (int o = 16; o > 0; o >>= 1) m = fmaxf(m, __shfl_xor(m, o, 32));

  float s = 0.f;
  for (int c = lane; c < kS; c += 32) s += __expf(p[c] - m);
#pragma unroll
  for (int o = 16; o > 0; o >>= 1) s += __shfl_xor(s, o, 32);

  const float inv = 1.0f / s;
  for (int c = lane; c < kS; c += 32) p[c] = __expf(p[c] - m) * inv;
}

// ---------------------------------------------------------------------------
// ctx_bf16[b,s,h*64+d] = sum_k P[bh,s,k] * V_head[k,d]
// P fp32 (converted in-kernel), V^T bf16 [B][D][S] (async tiles).
// grid: (S/TM, B*H)
// ---------------------------------------------------------------------------
__global__ __launch_bounds__(THREADS)
void attn_context_kernel(const float* __restrict__ attn,
                         const unsigned short* __restrict__ VT,
                         unsigned short* __restrict__ ctx) {
  __shared__ __align__(16) unsigned short lA[TM * LDSPAD];
  __shared__ __align__(16) unsigned short lB[TN * LDSPAD];

  const int tid  = threadIdx.x;
  const int lane = tid & 31;
  const int wave = tid >> 5;
  const int wm = (wave >> 1) * 32;
  const int wn = (wave & 1) * 32;
  const int qt = blockIdx.x * TM;
  const int bh = blockIdx.y;
  const int b = bh / kH, h = bh % kH;

  const float* P = attn + (size_t)bh * kS * kS;
  const unsigned short* Vh = VT + ((size_t)b * kD + h * kDK) * kS;

  const v8f zero = {0.f, 0.f, 0.f, 0.f, 0.f, 0.f, 0.f, 0.f};
  v8f acc[2][2];
  acc[0][0] = zero; acc[0][1] = zero; acc[1][0] = zero; acc[1][1] = zero;

  for (int kt = 0; kt < kS; kt += TK) {                 // 64 iterations
    // B tile: 64 d-rows x 16 bf16 (s = kt..kt+15), s-contiguous -> async, 1 chunk per thread
    {
      const int n = tid >> 1, ch = tid & 1;
      copy_b128_to_lds(Vh + (size_t)n * kS + kt + ch * 8,
                       &lB[n * LDSPAD + ch * 8]);
    }
    // A tile: 64 q-rows x 32 fp32 -> bf16, float4 -> uint2 LDS stores
#pragma unroll
    for (int e = 0; e < 4; ++e) {
      const int idx4 = tid + e * THREADS;               // 0..511
      const int r = idx4 >> 3, c4 = idx4 & 7;
      const float4 v =
          *(const float4*)(P + (size_t)(qt + r) * kS + kt + c4 * 4);
      uint2 o;
      o.x = pack_bf16x2(v.x, v.y);
      o.y = pack_bf16x2(v.z, v.w);
      *(uint2*)(&lA[r * LDSPAD + c4 * 4]) = o;
    }
    wait_async_lds();
    __syncthreads();

    v16bf af[2], bfr[2];
#pragma unroll
    for (int r = 0; r < 2; ++r)
      af[r] = load_a_frag(lA, wm + r * 16 + (lane & 15), lane);
#pragma unroll
    for (int c = 0; c < 2; ++c)
      bfr[c] = load_b_frag(lB, wn + c * 16 + (lane & 15), lane);
#pragma unroll
    for (int r = 0; r < 2; ++r)
#pragma unroll
      for (int c = 0; c < 2; ++c)
        acc[r][c] = wmma_bf16(af[r], bfr[c], acc[r][c]);
    __syncthreads();
  }

  const int g = lane >> 4, n15 = lane & 15;
#pragma unroll
  for (int r = 0; r < 2; ++r)
#pragma unroll
    for (int c = 0; c < 2; ++c) {
      const int col = wn + c * 16 + n15;                // dk in 0..63
#pragma unroll
      for (int vr = 0; vr < 8; ++vr) {
        const int row = qt + wm + r * 16 + vr + 8 * g;  // s within this b
        ctx[((size_t)b * kS + row) * kD + h * kDK + col] =
            f32_to_bf16(acc[r][c][vr]);
      }
    }
}

// ---------------------------------------------------------------------------
extern "C" void kernel_launch(void* const* d_in, const int* in_sizes, int n_in,
                              void* d_out, int out_size, void* d_ws, size_t ws_size,
                              hipStream_t stream) {
  const float* query = (const float*)d_in[0];
  const float* key   = (const float*)d_in[1];
  const float* value = (const float*)d_in[2];
  const int*   mask  = (const int*)d_in[3];
  const float* Wq = (const float*)d_in[4];  const float* bq = (const float*)d_in[5];
  const float* Wk = (const float*)d_in[6];  const float* bk = (const float*)d_in[7];
  const float* Wv = (const float*)d_in[8];  const float* bv = (const float*)d_in[9];
  const float* Wo = (const float*)d_in[10]; const float* bo = (const float*)d_in[11];

  float* out  = (float*)d_out;                          // [B,S,D] fp32
  float* attn = out + (size_t)kB * kS * kD;             // [B,H,S,S] fp32

  const size_t XN = (size_t)kB * kS * kD;               // 4,194,304
  const size_t WN = (size_t)kD * kD;                    // 1,048,576
  unsigned short* ws = (unsigned short*)d_ws;
  unsigned short* Xq  = ws;                             // bf16 inputs
  unsigned short* Xk  = Xq + XN;
  unsigned short* Xv  = Xk + XN;
  unsigned short* WqT = Xv + XN;                        // bf16 W^T
  unsigned short* WkT = WqT + WN;
  unsigned short* WvT = WkT + WN;
  unsigned short* WoT = WvT + WN;
  unsigned short* Qb  = WoT + WN;                       // bf16 projections
  unsigned short* Kb  = Qb + XN;
  unsigned short* VT  = Kb + XN;                        // bf16 V^T [B][D][S]
  unsigned short* Ctx = Xq;  // Xq dead after Q projection (stream-serialized)

  const dim3 blk(THREADS);

  // fp32 -> bf16 streaming conversions
  convert_bf16_kernel<<<(int)(XN / 4 / 256), 256, 0, stream>>>(
      (const float4*)query, (uint2*)Xq, (int)(XN / 4));
  convert_bf16_kernel<<<(int)(XN / 4 / 256), 256, 0, stream>>>(
      (const float4*)key, (uint2*)Xk, (int)(XN / 4));
  convert_bf16_kernel<<<(int)(XN / 4 / 256), 256, 0, stream>>>(
      (const float4*)value, (uint2*)Xv, (int)(XN / 4));

  const dim3 gT(kD / 32, kD / 32);
  transpose_convert_kernel<<<gT, dim3(32, 8), 0, stream>>>(Wq, WqT);
  transpose_convert_kernel<<<gT, dim3(32, 8), 0, stream>>>(Wk, WkT);
  transpose_convert_kernel<<<gT, dim3(32, 8), 0, stream>>>(Wv, WvT);
  transpose_convert_kernel<<<gT, dim3(32, 8), 0, stream>>>(Wo, WoT);

  // projections (bf16 WMMA, async LDS tiles)
  const dim3 gProj(kD / TN, (kB * kS) / TM);            // (16, 64)
  gemm_bf16_kernel<0><<<gProj, blk, 0, stream>>>(Xq, WqT, bq, Qb, kD, kD, kD, kD);
  gemm_bf16_kernel<0><<<gProj, blk, 0, stream>>>(Xk, WkT, bk, Kb, kD, kD, kD, kD);
  gemm_bf16_kernel<1><<<gProj, blk, 0, stream>>>(Xv, WvT, bv, VT, kD, kD, kD, kD);

  const dim3 gScore(kS / TN, kS / TM, kB * kH);         // (32, 32, 32)
  attn_scores_kernel<<<gScore, blk, 0, stream>>>(Qb, Kb, mask, attn);

  softmax_kernel<<<(kB * kH * kS) / 4, blk, 0, stream>>>(attn);

  const dim3 gCtx(kS / TM, kB * kH);                    // (32, 32)
  attn_context_kernel<<<gCtx, blk, 0, stream>>>(attn, VT, Ctx);

  gemm_bf16_kernel<2><<<gProj, blk, 0, stream>>>(Ctx, WoT, bo, out, kD, kD, kD, kD);
}
